// GraphSelfAttentionOld_6073083757195
// MI455X (gfx1250) — matrix operand order repaired and checked by reading, not compiled, not run on hardware
//
#include <hip/hip_runtime.h>
#include <hip/hip_bf16.h>

typedef _Float16 v16h __attribute__((ext_vector_type(16)));
typedef float    v8f  __attribute__((ext_vector_type(8)));

#define NUM_HEADS 16
#define SEQ 1024
#define BATCH 4
#define HIDDEN 1024
#define WCOLS 3328   // 16 heads * 208
#define VALSZ 64
#define NCT 208      // column tiles in proj (3328/16)

static __device__ __forceinline__ float redmax16(float v) {
#pragma unroll
  for (int m = 1; m < 16; m <<= 1) v = fmaxf(v, __shfl_xor(v, m, 32));
  return v;
}
static __device__ __forceinline__ float redsum16(float v) {
#pragma unroll
  for (int m = 1; m < 16; m <<= 1) v += __shfl_xor(v, m, 32);
  return v;
}

// ---------------------------------------------------------------------------
// One-time fp32 -> f16 conversion of H into WMMA *A-operand* lane order.
// Hf[((rt*32 + kk)*32 + lane)*16 + e] = H[rt*16 + (lane&15)][kk*32 + k_in(e)]
// ---------------------------------------------------------------------------
__global__ __launch_bounds__(256) void cvtH_kernel(
    const float* __restrict__ Hm, _Float16* __restrict__ Hf)
{
  const int idx = blockIdx.x * 256 + threadIdx.x;   // 0 .. 4M-1, row-major
  const int row = idx >> 10;
  const int k   = idx & 1023;
  const int rt = row >> 4, ln = row & 15;
  const int kk = k >> 5,   r  = k & 31;
  const int g16 = (r >> 3) & 1;                 // A layout inverse mapping
  const int e   = (r & 7) + ((r >> 4) << 3);
  const int lane = g16 * 16 + ln;
  Hf[(((size_t)(rt * 32 + kk) * 32 + lane) << 4) + e] = (_Float16)Hm[idx];
}

// ---------------------------------------------------------------------------
// One-time fp32 -> f16 conversion of W into WMMA *B-operand* lane order.
// Wf[((kk*208 + ct)*32 + g16*16 + ln)*16 + e] = W[kk*32 + 8*g16 + (e&7) + 16*(e>>3)][ct*16+ln]
// ---------------------------------------------------------------------------
__global__ __launch_bounds__(256) void cvtW_kernel(
    const float* __restrict__ Wm, _Float16* __restrict__ Wf)
{
  const int col = blockIdx.x * 256 + threadIdx.x;   // 0..3327
  const int k   = blockIdx.y;                       // 0..1023
  const int kk = k >> 5, r = k & 31;
  const int g16 = (r >> 3) & 1, e = (r & 7) + ((r >> 4) << 3); // B layout: table order
  const int ct = col >> 4, ln = col & 15;
  const size_t idx = ((size_t)(kk * NCT + ct) * 32 + g16 * 16 + ln) * 16 + e;
  Wf[idx] = (_Float16)Wm[(size_t)k * WCOLS + col];
}

// ---------------------------------------------------------------------------
// Kernel 1: H2 = H @ W via f16 WMMA. Operands pre-laid-out: hot loop is just
// two 32B vector loads + one v_wmma per K-step. One wave per 16x16 tile.
// Q/K/R scattered row-major [b][h][s][c]; V scattered TRANSPOSED [b][h][v][s]
// so the attention PV B-operand becomes a contiguous 32B vector load.
// ---------------------------------------------------------------------------
__global__ __launch_bounds__(128) void proj_kernel(
    const _Float16* __restrict__ Hf, const _Float16* __restrict__ Wf,
    _Float16* __restrict__ Qws, _Float16* __restrict__ Kws,
    _Float16* __restrict__ Vt, _Float16* __restrict__ Rws)
{
  const int lane = threadIdx.x & 31;
  const int wv   = threadIdx.x >> 5;
  const int ct   = __builtin_amdgcn_readfirstlane(blockIdx.x * 4 + wv); // 0..207
  const int rt   = blockIdx.y;                                         // 0..255
  const int ln   = lane & 15;
  const int g16  = lane >> 4;

  const v16h* __restrict__ Av = (const v16h*)Hf + (size_t)rt * 1024 + lane;
  const v16h* __restrict__ Bv = (const v16h*)Wf + (size_t)ct * 32 + g16 * 16 + ln;

  v8f c = {};
#pragma unroll 1
  for (int kk = 0; kk < 32; ++kk) {
    const v16h a  = Av[(size_t)kk * 32];
    const v16h bm = Bv[(size_t)kk * (NCT * 32)];
    c = __builtin_amdgcn_wmma_f32_16x16x32_f16(false, a, false, bm,
                                               (short)0, c, false, false);
  }

  // wave-uniform scatter: col tile -> (head, Q/K/V/R panel)
  const int h  = ct / 13;
  const int t0 = (ct % 13) * 16;          // uniform; 16-wide tiles don't straddle
  const int b  = rt >> 6;
  const int s0 = (rt & 63) * 16;
  const int bh = b * NUM_HEADS + h;
  const size_t bhbase = (size_t)bh * SEQ + s0;

  if (t0 < 128) {                         // Q or K, row-major [bh][s][64]
    _Float16* dst = (t0 < 64) ? (Qws + t0) : (Kws + (t0 - 64));
#pragma unroll
    for (int g = 0; g < 8; ++g)
      dst[(bhbase + g + 8 * g16) * 64 + ln] = (_Float16)c[g];
  } else if (t0 < 192) {                  // V, transposed [bh][v][s]
    _Float16* dst = Vt + ((size_t)bh * 64 + (t0 - 128) + ln) * SEQ + s0;
#pragma unroll
    for (int g = 0; g < 8; ++g)
      dst[g + 8 * g16] = (_Float16)c[g];
  } else {                                // R, row-major [bh][s][16]
#pragma unroll
    for (int g = 0; g < 8; ++g)
      Rws[(bhbase + g + 8 * g16) * 16 + ln] = (_Float16)c[g];
  }
}

// ---------------------------------------------------------------------------
// Kernel 2: fused RD + flash attention. Block = 16 waves = one (b, 16-query
// tile) for ALL heads, so each D tile is read from HBM exactly once (NT hint
// keeps the 256MB D stream from evicting the L2-resident Q/K/V/R panels).
// Wave w: RD worker for query i0+w (A-rows = heads), attention wave for head w.
// ---------------------------------------------------------------------------
__global__ __launch_bounds__(512) void attn_kernel(
    const float* __restrict__ Dm, const float* __restrict__ am,
    const _Float16* __restrict__ Qws, const _Float16* __restrict__ Kws,
    const _Float16* __restrict__ Vt, const _Float16* __restrict__ Rws,
    float* __restrict__ out)
{
  __shared__ __align__(16) float    RDlds[16 * 16 * 16];  // [i][h][j] fp32
  __shared__ __align__(16) _Float16 Plds[16 * 16 * 16];   // [wave][i][j] f16

  const int tid  = threadIdx.x;
  const int wv   = tid >> 5;          // head id (attention) / query offset (RD)
  const int lane = tid & 31;
  const int ln   = lane & 15;
  const int g16  = lane >> 4;

  const int b  = blockIdx.x >> 6;
  const int i0 = (blockIdx.x & 63) * 16;

  const float scale = 0.11180339887498948f;  // 1/sqrt(64+16)

  // Q as A operand (head wv, rows = queries), loop invariant, K=64 = 2 chunks
  const _Float16* qrow = Qws + ((size_t)(b * NUM_HEADS + wv) * SEQ + i0 + ln) * 64;
  v16h qa0, qa1;
#pragma unroll
  for (int e = 0; e < 8; ++e) {
    qa0[e]     = qrow[g16 * 8 + e];
    qa0[e + 8] = qrow[16 + g16 * 8 + e];
    qa1[e]     = qrow[32 + g16 * 8 + e];
    qa1[e + 8] = qrow[48 + g16 * 8 + e];
  }

  // R as A operand for RD matmul: rows = heads (lane ln), i = i0+wv fixed.
  // K padded 16->32: lanes 16..31 (k>=16) stay zero.
  const _Float16* rrow = Rws + ((size_t)(b * NUM_HEADS + ln) * SEQ + i0 + wv) * 16;
  v16h ra = {};
#pragma unroll
  for (int e = 0; e < 16; ++e) ra[e] = g16 ? (_Float16)0 : rrow[e];

  const float* dbase = Dm + ((size_t)b * SEQ + (i0 + wv)) * (16 * SEQ);
  const _Float16* kpan  = Kws + (size_t)(b * NUM_HEADS + wv) * SEQ * 64;
  const _Float16* vtpan = Vt  + (size_t)(b * NUM_HEADS + wv) * 64 * SEQ;

  float mrow[8], lrow[8];
  v8f o0 = {}, o1 = {}, o2 = {}, o3 = {};
#pragma unroll
  for (int g = 0; g < 8; ++g) { mrow[g] = -1e30f; lrow[g] = 0.0f; }

#pragma unroll 1
  for (int j0 = 0; j0 < SEQ; j0 += 16) {
    __syncthreads();  // RDlds / Plds free from previous iteration

    // ---- RD_i = R_heads(16x16) @ D_i(16x16), i = i0+wv; one HBM pass over D.
    // B lanes 16..31 (k>=16) carry zeros; lanes 0..15 load coalesced j-rows.
    v16h db = {};
    if (g16 == 0) {
#pragma unroll
      for (int e = 0; e < 16; ++e)
        db[e] = (_Float16)__builtin_nontemporal_load(
                    &dbase[(size_t)e * SEQ + j0 + ln]);
      // prefetch next j-step's D rows (the only HBM-bound stream)
      if (j0 + 16 < SEQ) {
#pragma unroll
        for (int e = 0; e < 16; ++e)
          __builtin_prefetch(&dbase[(size_t)e * SEQ + j0 + 16 + ln], 0, 1);
      }
    }
    v8f crd = {};
    crd = __builtin_amdgcn_wmma_f32_16x16x32_f16(false, ra, false, db,
                                                 (short)0, crd, false, false);
#pragma unroll
    for (int g = 0; g < 8; ++g)
      RDlds[(wv * 16 + (g + 8 * g16)) * 16 + ln] = crd[g];  // [i][h=row][j]

    __syncthreads();

    // ---- scores for head wv: QK^T (K=64) + RD + mask ----
    v16h kb0, kb1;
    {
      const _Float16* kr = kpan + (size_t)(j0 + ln) * 64;  // B col ln = key j
#pragma unroll
      for (int e = 0; e < 16; ++e) {
        kb0[e] = kr[g16 * 8 + (e & 7) + ((e >> 3) << 4)];
        kb1[e] = kr[32 + g16 * 8 + (e & 7) + ((e >> 3) << 4)];
      }
    }
    v8f qk = {};
    qk = __builtin_amdgcn_wmma_f32_16x16x32_f16(false, qa0, false, kb0,
                                                (short)0, qk, false, false);
    qk = __builtin_amdgcn_wmma_f32_16x16x32_f16(false, qa1, false, kb1,
                                                (short)0, qk, false, false);

    const float mval = am[(size_t)b * SEQ + j0 + ln];

    float p[8];
#pragma unroll
    for (int g = 0; g < 8; ++g) {
      const int irow = g + 8 * g16;
      float s = scale * (qk[g] + RDlds[(irow * 16 + wv) * 16 + ln]) + mval;
      float rm    = redmax16(s);                 // row lives in a 16-lane half
      float mnew  = fmaxf(mrow[g], rm);
      float pv    = __expf(s - mnew);
      float rs    = redsum16(pv);
      float alpha = __expf(mrow[g] - mnew);
      lrow[g] = lrow[g] * alpha + rs;
      mrow[g] = mnew;
      o0[g] *= alpha; o1[g] *= alpha; o2[g] *= alpha; o3[g] *= alpha;
      p[g] = pv;
    }

    // ---- re-layout P (C layout -> A layout) through LDS, f16 ----
#pragma unroll
    for (int g = 0; g < 8; ++g)
      Plds[(wv * 16 + (g + 8 * g16)) * 16 + ln] = (_Float16)p[g];

    __syncthreads();

    v16h pa = {};  // K padded 16->32: lanes 16..31 zero
#pragma unroll
    for (int e = 0; e < 16; ++e)
      pa[e] = g16 ? (_Float16)0 : Plds[(wv * 16 + ln) * 16 + e];

    // ---- O += P @ V : 4 subtiles; V pre-transposed -> one 32B load each.
    // B lanes 16..31 stay zero (k>=16 padding). ----
    v16h vb0 = {}, vb1 = {}, vb2 = {}, vb3 = {};
    if (g16 == 0) {
      vb0 = *(const v16h*)(vtpan + (size_t)(ln)      * SEQ + j0);
      vb1 = *(const v16h*)(vtpan + (size_t)(16 + ln) * SEQ + j0);
      vb2 = *(const v16h*)(vtpan + (size_t)(32 + ln) * SEQ + j0);
      vb3 = *(const v16h*)(vtpan + (size_t)(48 + ln) * SEQ + j0);
    }
    o0 = __builtin_amdgcn_wmma_f32_16x16x32_f16(false, pa, false, vb0, (short)0, o0, false, false);
    o1 = __builtin_amdgcn_wmma_f32_16x16x32_f16(false, pa, false, vb1, (short)0, o1, false, false);
    o2 = __builtin_amdgcn_wmma_f32_16x16x32_f16(false, pa, false, vb2, (short)0, o2, false, false);
    o3 = __builtin_amdgcn_wmma_f32_16x16x32_f16(false, pa, false, vb3, (short)0, o3, false, false);
  }

  // ---- normalize and store: out[b][s][h*64+v], fp32 ----
#pragma unroll
  for (int g = 0; g < 8; ++g) {
    const int irow = i0 + g + 8 * g16;
    const float inv = 1.0f / lrow[g];
    float* orow = out + ((size_t)b * SEQ + irow) * (NUM_HEADS * VALSZ) + wv * 64 + ln;
    orow[0]  = o0[g] * inv;
    orow[16] = o1[g] * inv;
    orow[32] = o2[g] * inv;
    orow[48] = o3[g] * inv;
  }
}

extern "C" void kernel_launch(void* const* d_in, const int* in_sizes, int n_in,
                              void* d_out, int out_size, void* d_ws, size_t ws_size,
                              hipStream_t stream) {
  const float* Hm = (const float*)d_in[0];          // (4,1024,1024)
  const float* Dm = (const float*)d_in[1];          // (4,1,1024,16,1024)
  const float* am = (const float*)d_in[2];          // (4,1,1,1024)
  const float* Wm = (const float*)d_in[3];          // (1024,3328)
  float* out = (float*)d_out;                       // (4,1024,1024) fp32

  // f16 workspace: Q,K,Vt = 8 MB each, R = 2 MB, Hf = 8 MB, Wf = 6.5 MB
  _Float16* Qws = (_Float16*)d_ws;
  _Float16* Kws = Qws + (size_t)BATCH * NUM_HEADS * SEQ * 64;
  _Float16* Vt  = Kws + (size_t)BATCH * NUM_HEADS * SEQ * 64;
  _Float16* Rws = Vt  + (size_t)BATCH * NUM_HEADS * SEQ * 64;
  _Float16* Hf  = Rws + (size_t)BATCH * NUM_HEADS * SEQ * 16;
  _Float16* Wf  = Hf  + (size_t)BATCH * SEQ * HIDDEN;

  cvtH_kernel<<<(BATCH * SEQ * HIDDEN) / 256, 256, 0, stream>>>(Hm, Hf);
  dim3 gw(WCOLS / 256, HIDDEN);
  cvtW_kernel<<<gw, 256, 0, stream>>>(Wm, Wf);

  dim3 g1(52, 256);   // 52*4 waves = 208 col tiles, 256 row tiles
  proj_kernel<<<g1, 128, 0, stream>>>(Hf, Wf, Qws, Kws, Vt, Rws);
  attn_kernel<<<256, 512, 0, stream>>>(Dm, am, Qws, Kws, Vt, Rws, out);
}
